// VoxelResBackBone8x_21139829031361
// MI455X (gfx1250) — matrix operand order repaired and checked by reading, not compiled, not run on hardware
//
#include <hip/hip_runtime.h>

// ---------------------------------------------------------------------------
// VoxelResBackBone8x on MI455X (gfx1250), NHWC activations, all convs lowered
// to implicit GEMM on v_wmma_f32_16x16x32_bf16 (wave32). Workgroup = 8 waves;
// each wave owns a 16-site tile. The weight panel is staged k-chunk by k-chunk
// into LDS by the Tensor Data Mover (tensor_load_to_lds + s_wait_tensorcnt),
// and consumed via ds_load b128 A-fragment reads.
// Conv biases are dropped (BN mean-subtraction cancels them exactly).
// ---------------------------------------------------------------------------

typedef __attribute__((ext_vector_type(16))) __bf16 v16bf;
typedef __attribute__((ext_vector_type(8)))  __bf16 v8bf;
typedef __attribute__((ext_vector_type(8)))  float  v8f;
typedef __attribute__((ext_vector_type(4)))  unsigned int u32x4;
typedef __attribute__((ext_vector_type(8)))  int          i32x8;
typedef __attribute__((ext_vector_type(4)))  int          i32x4;

#define DEVFN __device__ __forceinline__

constexpr int   H_   = 160;
constexpr int   W_   = 140;
constexpr int   HW_  = H_ * W_;
constexpr float BN_EPS = 1e-3f;

DEVFN unsigned short f2bf(float f) {
  unsigned int u = __builtin_bit_cast(unsigned int, f);
  u += 0x7FFFu + ((u >> 16) & 1u);             // round-to-nearest-even
  return (unsigned short)(u >> 16);
}
DEVFN __bf16 us2bf(unsigned short u) { return __builtin_bit_cast(__bf16, u); }

// TDM: load a [rows][32] bf16 tile (row stride = Kpad elements) from `gaddr`
// into LDS at byte offset `ldsAddr`.  D# packing per CDNA5 ISA §8.3/8.4.
DEVFN void tdm_load_panel(unsigned long long gaddr, unsigned ldsAddr,
                          int rows, int kpad) {
  u32x4 g0;
  g0[0] = 1u;                                   // count=1 (valid descriptor)
  g0[1] = ldsAddr;                              // lds_addr (bytes)
  g0[2] = (unsigned)gaddr;                      // global_addr[31:0]
  g0[3] = (unsigned)((gaddr >> 32) & 0x01FFFFFFu) | 0x80000000u;  // [56:32]|type=2
  i32x8 g1;
  g1[0] = 1 << 16;                              // data_size=1 -> 2 bytes
  g1[1] = (kpad & 0xFFFF) << 16;                // tensor_dim0[15:0] @bit48
  g1[2] = ((kpad >> 16) & 0xFFFF) | ((rows & 0xFFFF) << 16);  // dim0 hi | dim1 lo
  g1[3] = ((rows >> 16) & 0xFFFF) | (32 << 16); // dim1 hi | tile_dim0=32
  g1[4] = rows & 0xFFFF;                        // tile_dim1=rows, tile_dim2=0
  g1[5] = kpad;                                 // tensor_dim0_stride[31:0]
  g1[6] = 0;                                    // stride0 hi | stride1 lo
  g1[7] = 0;                                    // stride1 hi
  i32x4 z4 = {0, 0, 0, 0};
#if __clang_major__ >= 23
  i32x8 z8 = {0, 0, 0, 0, 0, 0, 0, 0};
  __builtin_amdgcn_tensor_load_to_lds(g0, g1, z4, z4, z8, 0);
#else
  __builtin_amdgcn_tensor_load_to_lds(g0, g1, z4, z4, 0);
#endif
}

// --------------------------- scatter voxels (NHWC) ------------------------
__global__ void scatter_kernel(const float* __restrict__ feats,
                               const int* __restrict__ coords, int N,
                               float* __restrict__ xf,
                               unsigned short* __restrict__ xb,
                               float* __restrict__ mask) {
  int i = blockIdx.x * 256 + threadIdx.x;
  if (i >= N) return;
  int z = coords[i * 4 + 1], y = coords[i * 4 + 2], x = coords[i * 4 + 3];
  int s = (z * H_ + y) * W_ + x;
  mask[s] = 1.f;
#pragma unroll
  for (int c = 0; c < 4; ++c) {
    float v = feats[i * 4 + c];
    xf[(size_t)s * 4 + c] = v;
    xb[(size_t)s * 4 + c] = f2bf(v);
  }
}

// -------------------- weight repack: [O][I][T] -> bf16 [O][Kpad] ----------
// K ordering: k = tap*Cin + ci (Cin is a power of two on every layer).
__global__ void repack_kernel(const float* __restrict__ w,
                              unsigned short* __restrict__ dst,
                              int Cin, int l2c, int taps, int Ktot, int Kpad) {
  int k = blockIdx.x * 256 + threadIdx.x;
  if (k >= Kpad) return;
  int co = blockIdx.y;
  unsigned short v = 0;
  if (k < Ktot) {
    int tap = k >> l2c;
    int ci  = k & (Cin - 1);
    v = f2bf(w[(co * Cin + ci) * taps + tap]);
  }
  dst[(size_t)co * Kpad + k] = v;
}

// ------------------- implicit-GEMM conv via WMMA bf16 ---------------------
// 256-thread WG = 8 waves; wave w owns sites [blk*128 + w*16, +16).
// Weight k-chunk (Cout x 32 bf16) staged in LDS by the TDM each iteration.
template <int KD, int KH, int KW, bool VEC16, int MT>
__global__ void conv_wmma_kernel(const unsigned short* __restrict__ xb,
                                 const unsigned short* __restrict__ wt,
                                 float* __restrict__ out,
                                 int Cin, int l2c, int Ktot, int Kpad,
                                 int Din, int Dout, int sz,
                                 int pz, int py, int px) {
  constexpr int KHW  = KH * KW;
  constexpr int Cout = MT * 16;
  __shared__ unsigned short lw[Cout * 32];     // one k-chunk of the A panel

  const int tid  = threadIdx.x;                // 0..255
  const int lane = tid & 31;
  const int col  = lane & 15;                  // N within this wave's tile
  const int hi   = lane >> 4;
  const int wv   = tid >> 5;                   // wave id 0..7
  const int Sout = Dout * HW_;
  const int site = blockIdx.x * 128 + wv * 16 + col;

  const bool sval = site < Sout;
  const int  st   = sval ? site : 0;
  const int  oz = st / HW_;
  const int  r0 = st - oz * HW_;
  const int  oy = r0 / W_;
  const int  ox = r0 - oy * W_;
  const int  iz0 = oz * sz - pz, iy0 = oy - py, ix0 = ox - px;

  v8f acc[MT];
#pragma unroll
  for (int m = 0; m < MT; ++m) acc[m] = (v8f)0.f;

  // A (16-bit 16x32): lane m = col; element j <-> K = kbA+j (lanes<16 own
  // K 0..15, lanes>=16 own K 16..31).  B (32x16): lane n = col; element j <-> K = kbB+j.
  const int kbA = hi ? 16 : 0;
  const int kbB = hi ? 16 : 0;

  const unsigned ldsAddr = (unsigned)(size_t)(void*)lw;  // LDS aperture: low32

  for (int k0 = 0; k0 < Kpad; k0 += 32) {
    // ---- TDM stage of weight chunk [Cout][k0..k0+32) into LDS ----
    // wave-uniform branch: only wave 0 issues (TDM ignores EXEC).
    if (wv == 0) {
      unsigned long long ga =
          (unsigned long long)(size_t)wt + (size_t)k0 * 2;
      tdm_load_panel(ga, ldsAddr, Cout, Kpad);
      __builtin_amdgcn_s_wait_tensorcnt(0);
    }
    __syncthreads();

    // ---- B fragment (per wave, shared by all MT tiles) ----
    v16bf b;
    if constexpr (VEC16) {
      const int kb = k0 + kbB;                 // multiple of 16; Cin >= 16 =>
      uint4 blo = {0, 0, 0, 0};                // 16-element run is one tap,
      uint4 bhi = {0, 0, 0, 0};                // contiguous channels.
      if (sval && kb < Ktot) {
        int tap = kb >> l2c;
        int ci0 = kb & (Cin - 1);
        int dz = tap / KHW;
        int rr = tap - dz * KHW;
        int dy = rr / KW;
        int dx = rr - dy * KW;
        int iz = iz0 + dz, iy = iy0 + dy, ix = ix0 + dx;
        if ((unsigned)iz < (unsigned)Din && (unsigned)iy < (unsigned)H_ &&
            (unsigned)ix < (unsigned)W_) {
          const unsigned short* p =
              xb + (size_t)((iz * H_ + iy) * W_ + ix) * Cin + ci0;
          blo = *(const uint4*)p;
          bhi = *(const uint4*)(p + 8);
        }
      }
      union { uint4 u; __bf16 h[8]; } ulo, uhi;
      ulo.u = blo;
      uhi.u = bhi;
#pragma unroll
      for (int j = 0; j < 8; ++j) { b[j] = ulo.h[j]; b[8 + j] = uhi.h[j]; }
    } else {
#pragma unroll
      for (int j = 0; j < 16; ++j) {
        int k = k0 + kbB + j;
        unsigned short v = 0;
        if (sval && k < Ktot) {
          int tap = k >> l2c;
          int ci  = k & (Cin - 1);
          int dz = tap / KHW;
          int rr = tap - dz * KHW;
          int dy = rr / KW;
          int dx = rr - dy * KW;
          int iz = iz0 + dz, iy = iy0 + dy, ix = ix0 + dx;
          if ((unsigned)iz < (unsigned)Din && (unsigned)iy < (unsigned)H_ &&
              (unsigned)ix < (unsigned)W_)
            v = xb[(size_t)((iz * H_ + iy) * W_ + ix) * Cin + ci];
        }
        b[j] = us2bf(v);
      }
    }

    // ---- MT A fragments from LDS + WMMAs against the shared B ----
#pragma unroll
    for (int mt = 0; mt < MT; ++mt) {
      const unsigned short* lrow = lw + (mt * 16 + col) * 32 + kbA;
      v8bf alo = *(const v8bf*)(lrow);
      v8bf ahi = *(const v8bf*)(lrow + 8);
      v16bf a;
#pragma unroll
      for (int j = 0; j < 8; ++j) { a[j] = alo[j]; a[8 + j] = ahi[j]; }
      acc[mt] = __builtin_amdgcn_wmma_f32_16x16x32_bf16(
          false, a, false, b, (short)0, acc[mt], false, false);
    }
    __syncthreads();
  }

  if (sval) {
    // D layout: VGPR r <-> M = r + 8*hi, so each lane owns 8 contiguous
    // channels -> two b128 stores per tile (NHWC).
    float* op = out + (size_t)site * Cout + 8 * hi;
#pragma unroll
    for (int mt = 0; mt < MT; ++mt) {
      *(float4*)(op + mt * 16) =
          make_float4(acc[mt][0], acc[mt][1], acc[mt][2], acc[mt][3]);
      *(float4*)(op + mt * 16 + 4) =
          make_float4(acc[mt][4], acc[mt][5], acc[mt][6], acc[mt][7]);
    }
  }
}

// ------------------------------ masked BN ---------------------------------
__global__ void mask_count_kernel(const float* __restrict__ mask, int S,
                                  float* __restrict__ cnt) {
  __shared__ float sh[256];
  int s = blockIdx.x * 256 + threadIdx.x;
  sh[threadIdx.x] = (s < S) ? mask[s] : 0.f;
  __syncthreads();
  for (int o = 128; o; o >>= 1) {
    if (threadIdx.x < o) sh[threadIdx.x] += sh[threadIdx.x + o];
    __syncthreads();
  }
  if (threadIdx.x == 0) atomicAdd(cnt, sh[0]);
}

__global__ void bn_stats_kernel(const float* __restrict__ x,
                                const float* __restrict__ mask, int S, int C,
                                float* __restrict__ sum,
                                float* __restrict__ sumsq) {
  __shared__ float s1[256], s2[256];
  int c = blockIdx.y;
  int s = blockIdx.x * 256 + threadIdx.x;
  float v = 0.f;
  if (s < S) v = x[(size_t)s * C + c] * mask[s];
  s1[threadIdx.x] = v;
  s2[threadIdx.x] = v * v;
  __syncthreads();
  for (int o = 128; o; o >>= 1) {
    if (threadIdx.x < o) {
      s1[threadIdx.x] += s1[threadIdx.x + o];
      s2[threadIdx.x] += s2[threadIdx.x + o];
    }
    __syncthreads();
  }
  if (threadIdx.x == 0) {
    atomicAdd(sum + c, s1[0]);
    atomicAdd(sumsq + c, s2[0]);
  }
}

__global__ void bn_finalize_kernel(const float* __restrict__ sum,
                                   const float* __restrict__ sumsq,
                                   const float* __restrict__ cnt,
                                   const float* __restrict__ g,
                                   const float* __restrict__ bt, int C,
                                   float* __restrict__ sc,
                                   float* __restrict__ sh) {
  int c = threadIdx.x;
  if (c >= C) return;
  float n    = cnt[0];
  float mean = sum[c] / n;
  float var  = sumsq[c] / n - mean * mean;
  float inv  = rsqrtf(var + BN_EPS);
  float s    = g[c] * inv;
  sc[c] = s;
  sh[c] = bt[c] - mean * s;
}

// normalize + optional residual + ReLU + mask-zero; emits f32 and/or bf16.
// chw=1 writes the f32 result channel-major (final output layout).
__global__ void apply_kernel(const float* __restrict__ x,
                             const float* __restrict__ mask,
                             const float* __restrict__ sc,
                             const float* __restrict__ sh,
                             const float* res, float* outf,
                             unsigned short* outb, int S, int C, int chw) {
  int c = blockIdx.y;
  int s = blockIdx.x * 256 + threadIdx.x;
  if (s >= S) return;
  size_t idx = (size_t)s * C + c;
  float v = x[idx] * sc[c] + sh[c];
  if (res) v += res[idx];
  v = fmaxf(v, 0.f);
  v = (mask[s] > 0.f) ? v : 0.f;
  if (outf) outf[chw ? ((size_t)c * S + s) : idx] = v;
  if (outb) outb[idx] = f2bf(v);
}

// ----------------------- downsampled active mask --------------------------
template <int KD, int KH, int KW>
__global__ void downmask_kernel(const float* __restrict__ mi,
                                float* __restrict__ mo, int Din, int Dout,
                                int sz, int pz, int py, int px) {
  int so = blockIdx.x * 256 + threadIdx.x;
  int Sout = Dout * HW_;
  if (so >= Sout) return;
  int oz = so / HW_;
  int r  = so - oz * HW_;
  int oy = r / W_;
  int ox = r - oy * W_;
  int iz0 = oz * sz - pz, iy0 = oy - py, ix0 = ox - px;
  float any = 0.f;
#pragma unroll
  for (int dz = 0; dz < KD; ++dz)
#pragma unroll
    for (int dy = 0; dy < KH; ++dy)
#pragma unroll
      for (int dx = 0; dx < KW; ++dx) {
        int iz = iz0 + dz, iy = iy0 + dy, ix = ix0 + dx;
        if ((unsigned)iz < (unsigned)Din && (unsigned)iy < (unsigned)H_ &&
            (unsigned)ix < (unsigned)W_)
          any += mi[(iz * H_ + iy) * W_ + ix];
      }
  mo[so] = (any > 0.f) ? 1.f : 0.f;
}

// ---------------------------- host-side glue ------------------------------
struct ConvW { const unsigned short* wt; int Cout, Cin, l2c, Ktot, Kpad; };

typedef void (*ConvFn)(const ConvW&, const unsigned short*, float*, int, int,
                       int, int, int, int, hipStream_t);

template <int KD, int KH, int KW, bool V, int MT>
static void launch_conv(const ConvW& w, const unsigned short* xb, float* outp,
                        int Din, int Dout, int sz, int pz, int py, int px,
                        hipStream_t stream) {
  dim3 g((Dout * HW_ + 127) / 128);   // 8 waves/WG, 16 sites/wave
  conv_wmma_kernel<KD, KH, KW, V, MT><<<g, 256, 0, stream>>>(
      xb, w.wt, outp, w.Cin, w.l2c, w.Ktot, w.Kpad, Din, Dout, sz, pz, py, px);
}

extern "C" void kernel_launch(void* const* d_in, const int* in_sizes, int n_in,
                              void* d_out, int out_size, void* d_ws,
                              size_t ws_size, hipStream_t stream) {
  if (n_in < 82) return;  // expects insertion-order flattening of setup_inputs()

  const int S0 = 41 * HW_, S4 = 1 * HW_;
  const size_t CAP = 15052800;  // max sites*channels over all stages (32*S1)

  // ---- workspace layout (bump allocator, 256B aligned) ----
  char* base = (char*)d_ws;
  size_t off = 0;
  auto alloc = [&](size_t bytes) -> char* {
    char* p = base + off;
    off = (off + bytes + 255) & ~(size_t)255;
    return p;
  };
  float*          X     = (float*)alloc(CAP * 4);
  float*          T     = (float*)alloc(CAP * 4);
  unsigned short* Xb    = (unsigned short*)alloc(CAP * 2);
  unsigned short* Yb    = (unsigned short*)alloc(CAP * 2);
  float*          mA    = (float*)alloc((size_t)S0 * 4);
  float*          mB    = (float*)alloc((size_t)S0 * 4);
  float*          m4    = (float*)alloc((size_t)S4 * 4);
  unsigned short* Waren = (unsigned short*)alloc((size_t)3000000 * 2);
  float*          stats = (float*)alloc(1024 * 4);
  (void)ws_size;
  float* sumP = stats, *sqP = stats + 128, *cntP = stats + 256,
       * scP = stats + 288, *shP = stats + 416;

  // ---- weight repack helper ----
  size_t wOff = 0;
  auto repack = [&](int widx, int Cout, int Cin, int l2c, int taps) -> ConvW {
    int Ktot = taps * Cin;
    int Kpad = (Ktot + 31) & ~31;
    unsigned short* dst = Waren + wOff;
    wOff += (size_t)Cout * Kpad;
    dim3 g((Kpad + 255) / 256, Cout);
    repack_kernel<<<g, 256, 0, stream>>>((const float*)d_in[widx], dst, Cin,
                                         l2c, taps, Ktot, Kpad);
    ConvW w; w.wt = dst; w.Cout = Cout; w.Cin = Cin; w.l2c = l2c;
    w.Ktot = Ktot; w.Kpad = Kpad;
    return w;
  };

  auto runBN = [&](const float* xin, const float* mask, int C, int S,
                   int gIdx, int btIdx, const float* res, float* outf,
                   unsigned short* outb, int chw) {
    hipMemsetAsync(stats, 0, 272 * 4, stream);
    mask_count_kernel<<<(S + 255) / 256, 256, 0, stream>>>(mask, S, cntP);
    dim3 gs((S + 255) / 256, C);
    bn_stats_kernel<<<gs, 256, 0, stream>>>(xin, mask, S, C, sumP, sqP);
    bn_finalize_kernel<<<1, 128, 0, stream>>>(sumP, sqP, cntP,
                                              (const float*)d_in[gIdx],
                                              (const float*)d_in[btIdx], C,
                                              scP, shP);
    apply_kernel<<<gs, 256, 0, stream>>>(xin, mask, scP, shP, res, outf, outb,
                                         S, C, chw);
  };

  const float* curMask = mA;
  auto basicBlock = [&](int bi, int C, int l2c, int D, ConvFn fn) {
    ConvW w1 = repack(bi + 0, C, C, l2c, 27);
    ConvW w2 = repack(bi + 4, C, C, l2c, 27);
    int S = D * HW_;
    fn(w1, Xb, T, D, D, 1, 1, 1, 1, stream);
    runBN(T, curMask, C, S, bi + 2, bi + 3, nullptr, T, Yb, 0);   // y -> Yb
    fn(w2, Yb, T, D, D, 1, 1, 1, 1, stream);
    runBN(T, curMask, C, S, bi + 6, bi + 7, X, X, Xb, 0);         // + residual
  };
  auto down = [&](int wi, int Cin, int l2cin, int Cout, int Din, int Dout,
                  int pz, float* nm, ConvFn fn) {
    ConvW w = repack(wi, Cout, Cin, l2cin, 27);
    fn(w, Xb, T, Din, Dout, 2, pz, 1, 1, stream);
    downmask_kernel<3, 3, 3><<<(Dout * HW_ + 255) / 256, 256, 0, stream>>>(
        curMask, nm, Din, Dout, 2, pz, 1, 1);
    runBN(T, nm, Cout, Dout * HW_, wi + 1, wi + 2, nullptr, X, Xb, 0);
    curMask = nm;
  };

  // ---- input scatter ----
  const int Nvox = in_sizes[0] / 4;
  hipMemsetAsync(X, 0, (size_t)4 * S0 * 4, stream);
  hipMemsetAsync(Xb, 0, (size_t)4 * S0 * 2, stream);
  hipMemsetAsync(mA, 0, (size_t)S0 * 4, stream);
  scatter_kernel<<<(Nvox + 255) / 256, 256, 0, stream>>>(
      (const float*)d_in[0], (const int*)d_in[1], Nvox, X, Xb, mA);

  // ---- conv_input: 4->16 (scalar-gather path, Cin=4) ----
  {
    ConvW wi = repack(3, 16, 4, 2, 27);
    launch_conv<3, 3, 3, false, 1>(wi, Xb, T, 41, 41, 1, 1, 1, 1, stream);
    runBN(T, mA, 16, S0, 4, 5, nullptr, X, Xb, 0);
  }
  // ---- stage 1 (16ch, D=41) ----
  basicBlock(6, 16, 4, 41, &launch_conv<3, 3, 3, true, 1>);
  basicBlock(14, 16, 4, 41, &launch_conv<3, 3, 3, true, 1>);
  // ---- stage 2: down 16->32 (z 41->21, pad_z=1), blocks at D=21 ----
  down(22, 16, 4, 32, 41, 21, 1, mB, &launch_conv<3, 3, 3, true, 2>);
  basicBlock(25, 32, 5, 21, &launch_conv<3, 3, 3, true, 2>);
  basicBlock(33, 32, 5, 21, &launch_conv<3, 3, 3, true, 2>);
  // ---- stage 3: down 32->64 (z 21->10, pad_z=0), blocks at D=10 ----
  down(41, 32, 5, 64, 21, 10, 0, mA, &launch_conv<3, 3, 3, true, 4>);
  basicBlock(44, 64, 6, 10, &launch_conv<3, 3, 3, true, 4>);
  basicBlock(52, 64, 6, 10, &launch_conv<3, 3, 3, true, 4>);
  // ---- stage 4: down 64->128 (z 10->4, pad_z=0), blocks at D=4 ----
  down(60, 64, 6, 128, 10, 4, 0, mB, &launch_conv<3, 3, 3, true, 8>);
  basicBlock(63, 128, 7, 4, &launch_conv<3, 3, 3, true, 8>);
  basicBlock(71, 128, 7, 4, &launch_conv<3, 3, 3, true, 8>);
  // ---- conv_out: 128->128, k=(3,1,1), stride(2,1,1), pad 0, z 4->1 ----
  {
    ConvW wo = repack(79, 128, 128, 7, 3);
    launch_conv<3, 1, 1, true, 8>(wo, Xb, T, 4, 1, 2, 0, 0, 0, stream);
    downmask_kernel<3, 1, 1><<<(S4 + 255) / 256, 256, 0, stream>>>(
        curMask, m4, 4, 1, 2, 0, 0, 0);
    runBN(T, m4, 128, S4, 80, 81, nullptr, (float*)d_out, nullptr, 1);
  }
  (void)out_size;
}
